// GCN_30485677867455
// MI455X (gfx1250) — compile-verified, matrix operand order checked by reading, not run on hardware
//
#include <hip/hip_runtime.h>
#include <math.h>

// ---------------------------------------------------------------------------
// GCN 2-layer forward for MI455X (gfx1250, wave32, WMMA).
//
// Math reordering (exact, since propagation is linear):
//   layer1: h1 = relu( Ahat @ (x @ W1) + b1 )          (GEMM then 16-dim prop)
//   layer2: out = log_softmax( (Ahat @ h1) @ W2 + b2 )  (16-dim prop then GEMM)
// Both GEMMs use v_wmma_f32_16x16x32_bf16. Weights are pre-packed into the
// exact per-lane WMMA B-fragment layout so the GEMM inner loops issue only
// wide b128 loads.
// ---------------------------------------------------------------------------

#define NN 5000   // nodes == in-dim == classes
#define FH 16     // hidden dim
#define NC ((NN + 31) / 32)   // 157 K-chunks of 32 for GEMM1
#define MT ((NN + 15) / 16)   // 313 16-wide tiles

typedef __attribute__((ext_vector_type(16))) __bf16 v16bf;
typedef __attribute__((ext_vector_type(8)))  __bf16 v8bf;
typedef __attribute__((ext_vector_type(8)))  float  v8f;

// ---------------- small utility kernels ----------------

__global__ void k_zero_f32(float* __restrict__ p, int n) {
    int i = blockIdx.x * blockDim.x + threadIdx.x;
    if (i < n) p[i] = 0.0f;
}

__global__ void k_count_deg(const int* __restrict__ dst, float* __restrict__ deg, int E) {
    int i = blockIdx.x * blockDim.x + threadIdx.x;
    if (i < E) atomicAdd(&deg[dst[i]], 1.0f);
}

__global__ void k_inv_sqrt(const float* __restrict__ deg, float* __restrict__ inv) {
    int i = blockIdx.x * blockDim.x + threadIdx.x;
    if (i < NN) inv[i] = rsqrtf(deg[i] + 1.0f);   // +1 self loop; always > 0
}

__global__ void k_cvt_bf16(const float* __restrict__ in, __bf16* __restrict__ out, int n) {
    int i = blockIdx.x * blockDim.x + threadIdx.x;
    if (i < n) out[i] = (__bf16)in[i];
}

__global__ void k_bias_relu(float* __restrict__ p, const float* __restrict__ b) {
    int i = blockIdx.x * blockDim.x + threadIdx.x;
    if (i < NN * FH) p[i] = fmaxf(p[i] + b[i & (FH - 1)], 0.0f);
}

// ---------------- weight pre-packing into WMMA B-fragment layout ------------
// 16-bit B layout (mirror of A): lane col = lane&15; kb = (lane>>4)*8;
//   fragment elem 0..7  -> K = kb+0..kb+7
//   fragment elem 8..15 -> K = kb+16..kb+23

// W1 [NN,16] f32 -> w1p: for chunk c, lane l: 16 contiguous bf16 (32B/lane)
__global__ void k_pack_w1(const float* __restrict__ W1, __bf16* __restrict__ w1p) {
    int t = blockIdx.x * blockDim.x + threadIdx.x;
    if (t >= NC * 32 * 16) return;
    int elem = t & 15;
    int lane = (t >> 4) & 31;
    int c    = t >> 9;                 // 32*16 = 512 per chunk
    int kb   = (lane >> 4) * 8;
    int col  = lane & 15;
    int K    = c * 32 + kb + ((elem < 8) ? elem : (8 + elem));  // +16 for hi half
    w1p[t] = (K < NN) ? (__bf16)W1[K * FH + col] : (__bf16)0.0f;
}

// W2 [16,NN] f32 -> w2p: for ntile nt, lane l: 8 contiguous bf16 (16B/lane)
// (K = 16 padded to 32 in the GEMM; only elems 0..7 (K = 0..15) are real)
__global__ void k_pack_w2(const float* __restrict__ W2, __bf16* __restrict__ w2p) {
    int t = blockIdx.x * blockDim.x + threadIdx.x;
    if (t >= MT * 32 * 8) return;
    int j    = t & 7;
    int lane = (t >> 3) & 31;
    int nt   = t >> 8;                 // 32*8 = 256 per tile
    int kb   = (lane >> 4) * 8;
    int col  = nt * 16 + (lane & 15);
    w2p[t] = (col < NN) ? (__bf16)W2[(size_t)(kb + j) * NN + col] : (__bf16)0.0f;
}

// ---------------- normalized propagation: hout[d] += inv[s]*inv[d]*hin[s] ---

__global__ void k_propagate(const int* __restrict__ src, const int* __restrict__ dst,
                            const float* __restrict__ inv,
                            const float* __restrict__ hin, float* __restrict__ hout,
                            int E) {
    int t = blockIdx.x * blockDim.x + threadIdx.x;
    int e = t >> 4;           // 16 lanes per edge: one per feature
    int f = t & 15;
    if (e >= E + NN) return;
    int s, d;
    if (e < E) { s = src[e]; d = dst[e]; }
    else       { s = d = e - E; }        // self loops
    float c = inv[s] * inv[d];
    atomicAdd(&hout[d * FH + f], c * hin[s * FH + f]);
}

// ---------------- GEMM1: h[NN,16] = x[NN,NN] @ W1[NN,16]  (bf16 WMMA) -------
// One wave per 16-row tile of h; K loop in chunks of 32.
// A (16-bit, 16x32): lane row = lane&15; kb = (lane>>4)*8;
//   elems 0..7 -> K = kb+0..7, elems 8..15 -> K = kb+16..23.

__global__ __launch_bounds__(128) void k_gemm1(const float* __restrict__ x,
                                               const __bf16* __restrict__ w1p,
                                               float* __restrict__ h) {
    const int wave = threadIdx.x >> 5;
    const int lane = threadIdx.x & 31;
    const int mtile = blockIdx.x * 4 + wave;
    if (mtile * 16 >= NN) return;                  // wave-uniform exit
    const int row = mtile * 16 + (lane & 15);
    const int rclamp = row < NN ? row : NN - 1;    // clamp; clamped rows never stored
    const float* __restrict__ xrow = x + (size_t)rclamp * NN;
    const int kb  = (lane >> 4) * 8;               // 0 or 8
    const int col = lane & 15;

    v8f acc = {};
    // full K=32 chunks: explicit b128 loads for the x stream (HBM-bound)
    for (int c = 0; c < NC - 1; ++c) {
        const float4* __restrict__ ap = (const float4*)(xrow + c * 32 + kb);
        float f[16];
        *(float4*)&f[0]  = ap[0];
        *(float4*)&f[4]  = ap[1];
        *(float4*)&f[8]  = ap[4];   // +16 floats
        *(float4*)&f[12] = ap[5];
        v16bf a;
#pragma unroll
        for (int j = 0; j < 16; ++j) a[j] = (__bf16)f[j];
        v16bf b = *(const v16bf*)(w1p + ((size_t)c * 32 + lane) * 16);
        acc = __builtin_amdgcn_wmma_f32_16x16x32_bf16(false, a, false, b,
                                                      (short)0, acc, false, false);
    }
    // tail chunk (NN % 32 == 8): zero-pad out-of-range K in A; B pack is padded
    {
        const int c = NC - 1;
        const int k0 = c * 32;
        v16bf a = {};
#pragma unroll
        for (int j = 0; j < 8; ++j) {
            int kA = k0 + kb + j, kB = kA + 16;
            if (kA < NN) a[j]     = (__bf16)xrow[kA];
            if (kB < NN) a[8 + j] = (__bf16)xrow[kB];
        }
        v16bf b = *(const v16bf*)(w1p + ((size_t)c * 32 + lane) * 16);
        acc = __builtin_amdgcn_wmma_f32_16x16x32_bf16(false, a, false, b,
                                                      (short)0, acc, false, false);
    }
    // C layout: VGPR r -> M = r + (lane>=16 ? 8 : 0), N = lane&15
    const int rbase = mtile * 16 + ((lane >> 4) << 3);
#pragma unroll
    for (int r = 0; r < 8; ++r) {
        int rr = rbase + r;
        if (rr < NN) h[rr * FH + col] = acc[r];
    }
}

// ---------------- GEMM2: out[NN,NN] = q[NN,16] @ W2[16,NN] + b2 -------------
// K = 16 padded to 32 (upper half of fragments zero): ONE wmma per 16x16 tile.

__global__ __launch_bounds__(128) void k_gemm2(const __bf16* __restrict__ qbf,
                                               const __bf16* __restrict__ w2p,
                                               const float* __restrict__ b2,
                                               float* __restrict__ out) {
    const int wave = threadIdx.x >> 5;
    const int lane = threadIdx.x & 31;
    const int mtile = blockIdx.x * 4 + wave;
    const int ntile = blockIdx.y;
    if (mtile * 16 >= NN) return;                  // wave-uniform exit
    const int mrow = mtile * 16 + (lane & 15);
    const int mrc  = mrow < NN ? mrow : NN - 1;
    const int kb   = (lane >> 4) * 8;              // 0 or 8
    const int col  = ntile * 16 + (lane & 15);
    const int colc = col < NN ? col : NN - 1;

    // single b128 load each for A and B fragments
    v8bf av = *(const v8bf*)(qbf + mrc * FH + kb);
    v8bf bv = *(const v8bf*)(w2p + ((size_t)ntile * 32 + lane) * 8);
    v16bf a = {}, b = {};                          // elems 8..15 (K>=16) stay 0
#pragma unroll
    for (int j = 0; j < 8; ++j) { a[j] = av[j]; b[j] = bv[j]; }

    v8f acc = {};
    acc = __builtin_amdgcn_wmma_f32_16x16x32_bf16(false, a, false, b,
                                                  (short)0, acc, false, false);

    const int   rbase = mtile * 16 + ((lane >> 4) << 3);
    const float bias  = b2[colc];
#pragma unroll
    for (int r = 0; r < 8; ++r) {
        int rr = rbase + r;
        if (rr < NN && col < NN) out[(size_t)rr * NN + col] = acc[r] + bias;
    }
}

// ---------------- row-wise log_softmax (in place on d_out) ------------------

__global__ __launch_bounds__(256) void k_log_softmax(float* __restrict__ out) {
    const int row = blockIdx.x;
    float* __restrict__ p = out + (size_t)row * NN;
    __shared__ float red[256];
    const int tid = threadIdx.x;

    float m = -INFINITY;
    for (int i = tid; i < NN; i += 256) m = fmaxf(m, p[i]);
    red[tid] = m; __syncthreads();
    for (int s = 128; s > 0; s >>= 1) {
        if (tid < s) red[tid] = fmaxf(red[tid], red[tid + s]);
        __syncthreads();
    }
    m = red[0]; __syncthreads();

    float sum = 0.0f;
    for (int i = tid; i < NN; i += 256) sum += expf(p[i] - m);
    red[tid] = sum; __syncthreads();
    for (int s = 128; s > 0; s >>= 1) {
        if (tid < s) red[tid] += red[tid + s];
        __syncthreads();
    }
    const float lse = m + logf(red[0]);

    for (int i = tid; i < NN; i += 256) p[i] -= lse;
}

// ---------------------------------------------------------------------------

extern "C" void kernel_launch(void* const* d_in, const int* in_sizes, int n_in,
                              void* d_out, int out_size, void* d_ws, size_t ws_size,
                              hipStream_t stream) {
    const float* x   = (const float*)d_in[0];
    const int*   src = (const int*)  d_in[1];
    const int*   dst = (const int*)  d_in[2];
    const float* W1  = (const float*)d_in[3];
    const float* b1  = (const float*)d_in[4];
    const float* W2  = (const float*)d_in[5];
    const float* b2  = (const float*)d_in[6];
    float*       out = (float*)d_out;
    const int E = in_sizes[1];

    // carve workspace (512B aligned slices)
    char* w = (char*)d_ws;
    auto carve = [&](size_t bytes) { void* p = w; w += (bytes + 511) & ~(size_t)511; return p; };
    float*  h1   = (float*) carve(NN * FH * sizeof(float));     // x@W1
    float*  p1   = (float*) carve(NN * FH * sizeof(float));     // Ahat h1 (+b1, relu)
    float*  q    = (float*) carve(NN * FH * sizeof(float));     // Ahat p1
    float*  deg  = (float*) carve(NN * sizeof(float));
    float*  inv  = (float*) carve(NN * sizeof(float));
    __bf16* w1p  = (__bf16*)carve((size_t)NC * 32 * 16 * sizeof(__bf16)); // packed B frags
    __bf16* w2p  = (__bf16*)carve((size_t)MT * 32 * 8  * sizeof(__bf16)); // packed B frags
    __bf16* qbf  = (__bf16*)carve(NN * FH * sizeof(__bf16));

    const int TB = 256;
    const int mblocks = (MT + 3) / 4;             // 79 blocks of 4 waves

    // zero accumulators (ws is poisoned; atomics need clean zeros each call)
    k_zero_f32<<<(NN + TB - 1) / TB, TB, 0, stream>>>(deg, NN);
    k_zero_f32<<<(NN * FH + TB - 1) / TB, TB, 0, stream>>>(p1, NN * FH);
    k_zero_f32<<<(NN * FH + TB - 1) / TB, TB, 0, stream>>>(q, NN * FH);

    // degree / normalization
    k_count_deg<<<(E + TB - 1) / TB, TB, 0, stream>>>(dst, deg, E);
    k_inv_sqrt<<<(NN + TB - 1) / TB, TB, 0, stream>>>(deg, inv);

    // pre-pack weights into WMMA B-fragment layout (bf16, zero-padded)
    k_pack_w1<<<(NC * 32 * 16 + TB - 1) / TB, TB, 0, stream>>>(W1, w1p);
    k_pack_w2<<<(MT * 32 * 8  + TB - 1) / TB, TB, 0, stream>>>(W2, w2p);

    // layer 1
    k_gemm1<<<mblocks, 128, 0, stream>>>(x, w1p, h1);
    {
        int tot = (E + NN) * 16;
        k_propagate<<<(tot + TB - 1) / TB, TB, 0, stream>>>(src, dst, inv, h1, p1, E);
    }
    k_bias_relu<<<(NN * FH + TB - 1) / TB, TB, 0, stream>>>(p1, b1);

    // layer 2 (propagate first — Ahat(hW2) == (Ahat h)W2)
    {
        int tot = (E + NN) * 16;
        k_propagate<<<(tot + TB - 1) / TB, TB, 0, stream>>>(src, dst, inv, p1, q, E);
    }
    k_cvt_bf16<<<(NN * FH + TB - 1) / TB, TB, 0, stream>>>(q, qbf, NN * FH);
    k_gemm2<<<dim3(mblocks, MT), 128, 0, stream>>>(qbf, w2p, b2, out);

    // log_softmax rows (second read of out hits 192MB L2)
    k_log_softmax<<<NN, 256, 0, stream>>>(out);
}
